// RelMultiHeadDotProductAttention_22814866276585
// MI455X (gfx1250) — matrix operand order repaired and checked by reading, not compiled, not run on hardware
//
#include <hip/hip_runtime.h>

// ---------------------------------------------------------------------------
// CDNA5 (gfx1250) wave32 WMMA helpers
// ---------------------------------------------------------------------------
typedef __attribute__((ext_vector_type(16))) __bf16 v16bf;
typedef __attribute__((ext_vector_type(8)))  float  v8f;
typedef __attribute__((ext_vector_type(4))) unsigned int v4u;
typedef __attribute__((ext_vector_type(8))) int v8i_t;
typedef __attribute__((ext_vector_type(4))) int v4i_t;

union Frag32B { v16bf v; uint4 u[2]; };

// Hardware convert (v_cvt_*_bf16_f32, RNE) instead of software rounding.
__device__ __forceinline__ __bf16 to_bf16(float f) { return (__bf16)f; }

#define WMMA_BF16(a, b, c) \
  __builtin_amdgcn_wmma_f32_16x16x32_bf16(false, (a), false, (b), (short)0, (c), false, false)

// A fragment (16x32 bf16, MxK) from a row-major LDS tile.
// ISA layout: lane m in [0,16): K = 0..7 / 16..23 ; lane m+16: K = 8..15 / 24..31.
// pitch must be a multiple of 8 elements so both 16B loads are aligned.
__device__ __forceinline__ v16bf lds_frag_A(const __bf16* base, int pitch, int k0) {
  const int lane = threadIdx.x & 31;
  const int m = lane & 15, hf = lane >> 4;
  const __bf16* p = base + m * pitch + k0 + hf * 8;
  Frag32B f;
  f.u[0] = *(const uint4*)p;          // K = hf*8 .. hf*8+7
  f.u[1] = *(const uint4*)(p + 16);   // K = 16+hf*8 .. 16+hf*8+7
  return f.v;
}

// B fragment (32x16 bf16, KxN) where B[k][n] = tile[n][k] (tile row-major in LDS).
// ISA layout: lanes 0-15 hold col n with K=0..7 / 16..23, lanes 16-31 hold K=8..15 / 24..31.
__device__ __forceinline__ v16bf lds_frag_BT(const __bf16* base, int pitch, int n0, int k0) {
  const int lane = threadIdx.x & 31;
  const int n = lane & 15, kh = lane >> 4;
  const __bf16* p = base + (n0 + n) * pitch + k0 + kh * 8;
  Frag32B f;
  f.u[0] = *(const uint4*)p;
  f.u[1] = *(const uint4*)(p + 16);
  return f.v;
}

// B fragment (32x16) gathered from global f32 row-major B[k][n]; per element the
// 32 lanes read two 64B-contiguous rows -> coalesced.
__device__ __forceinline__ v16bf glob_frag_B(const float* B, int ldb, int k0, int n0) {
  const int lane = threadIdx.x & 31;
  const int n = lane & 15, kh = lane >> 4;
  const float* p = B + (size_t)(k0 + kh * 8) * ldb + n0 + n;
  v16bf r;
#pragma unroll
  for (int i = 0; i < 16; ++i) r[i] = to_bf16(p[(size_t)(i + (i & 8)) * ldb]);
  return r;
}

// ---------------------------------------------------------------------------
// Tensor Data Mover: 2D tile load Global(bf16) -> LDS with row padding.
// Descriptor layout per CDNA5 ISA ch.8 (group0 128b, group1 256b).
// ---------------------------------------------------------------------------
#if __has_builtin(__builtin_amdgcn_tensor_load_to_lds)
#define HAS_TDM 1
#else
#define HAS_TDM 0
#endif

__device__ __forceinline__ void tdm_load_2d_bf16(unsigned lds_byte_addr, const void* gsrc,
                                                 unsigned tensor_d0, unsigned tensor_d1,
                                                 unsigned tile_d0, unsigned tile_d1,
                                                 unsigned long long stride0_elems) {
#if HAS_TDM
  unsigned long long ga = (unsigned long long)(size_t)gsrc;
  v4u g0;
  g0[0] = 1u;                                            // count=1, user mode
  g0[1] = lds_byte_addr;                                 // D#.lds_addr
  g0[2] = (unsigned)(ga & 0xFFFFFFFFu);                  // global_addr[31:0]
  g0[3] = (unsigned)((ga >> 32) & 0x1FFFFFFu) | (2u << 30); // addr[56:32] | type=2
  // group1: data_size=1 (2B), pad_enable, pad_interval=32 DWORDs (code 4),
  // pad_amount=4 DWORDs (code 3)  -> LDS row pitch = 128B + 16B = 144B (72 elems)
  unsigned w[8];
  w[0] = (1u << 16) | (1u << 20) | (4u << 22) | (3u << 25);
  w[1] = (tensor_d0 & 0xFFFFu) << 16;                    // tensor_dim0[15:0] @ bit48
  w[2] = (tensor_d0 >> 16) | ((tensor_d1 & 0xFFFFu) << 16);
  w[3] = (tensor_d1 >> 16) | ((tile_d0 & 0xFFFFu) << 16);
  w[4] = (tile_d1 & 0xFFFFu);                            // tile_dim2 = 0 (2D)
  w[5] = (unsigned)(stride0_elems & 0xFFFFFFFFu);
  w[6] = (unsigned)((stride0_elems >> 32) & 0xFFFFu);    // dim1_stride = 0 (2D)
  w[7] = 0u;
  v8i_t g1;
#pragma unroll
  for (int i = 0; i < 8; ++i) g1[i] = (int)w[i];
  v4i_t z4 = {0, 0, 0, 0};
  v8i_t z8 = {0, 0, 0, 0, 0, 0, 0, 0};
  __builtin_amdgcn_tensor_load_to_lds(g0, g1, z4, z4, z8, 0);
#else
  (void)lds_byte_addr; (void)gsrc; (void)tensor_d0; (void)tensor_d1;
  (void)tile_d0; (void)tile_d1; (void)stride0_elems;
#endif
}

__device__ __forceinline__ unsigned lds_addr_of(const void* p) {
  // generic (flat) LDS pointer: low 32 bits are the LDS byte address
  return (unsigned)(unsigned long long)(size_t)p;
}

// ---------------------------------------------------------------------------
// Generic C[M,N] = A[M,K] (f32) x B[K,N] (f32) + bias[N]; bf16 WMMA;
// OutT = float or __bf16. Block 256 thr (8 waves), tile 32 x 128:
// each wave owns a 32x16 column strip -> 2 WMMA per K-step reusing ONE
// B fragment (halves global gather + convert cost per matrix op).
// ---------------------------------------------------------------------------
template <typename OutT>
__global__ __launch_bounds__(256)
void gemm_bf16_wmma(const float* __restrict__ A, const float* __restrict__ Bm,
                    const float* __restrict__ bias, OutT* __restrict__ C,
                    int M, int N, int K) {
  __shared__ __bf16 aT[32 * 40];                 // 32 x 32 A panel, pitch 40 (80B rows)
  const int m0 = blockIdx.x * 32;
  const int n0 = blockIdx.y * 128 + (threadIdx.x >> 5) * 16;
  v8f acc0 = {}, acc1 = {};
  for (int k0 = 0; k0 < K; k0 += 32) {
    for (int idx = threadIdx.x; idx < 32 * 32; idx += 256) {
      int r = idx >> 5, c = idx & 31;
      aT[r * 40 + c] = to_bf16(A[(size_t)(m0 + r) * K + k0 + c]);
    }
    if (k0 + 32 < K)   // hint next B panel toward L2 (global_prefetch_b8)
      __builtin_prefetch(Bm + (size_t)(k0 + 32) * N + n0, 0, 1);
    __syncthreads();
    v16bf b  = glob_frag_B(Bm, N, k0, n0);       // shared by both M tiles
    v16bf a0 = lds_frag_A(aT, 40, 0);
    v16bf a1 = lds_frag_A(aT + 16 * 40, 40, 0);
    acc0 = WMMA_BF16(a0, b, acc0);
    acc1 = WMMA_BF16(a1, b, acc1);
    __syncthreads();
  }
  const int lane = threadIdx.x & 31;
  const int n = lane & 15, hf = lane >> 4;
  const float bv = bias ? bias[n0 + n] : 0.0f;
#pragma unroll
  for (int r = 0; r < 8; ++r) {
    int row = m0 + r + hf * 8;                   // C layout: VGPR r -> M = r + 8*(lane/16)
    C[(size_t)row * N + n0 + n]        = (OutT)(acc0[r] + bv);
    C[(size_t)(row + 16) * N + n0 + n] = (OutT)(acc1[r] + bv);
  }
}

// ---------------------------------------------------------------------------
// Fused Transformer-XL relative attention (flash-style, WMMA bf16, TDM staging).
// Grid: (S/64, H, B); block 256 threads = 8 waves; one 64-row q tile per block.
// ---------------------------------------------------------------------------
constexpr int ATTN_S  = 2048;
constexpr int ATTN_HD = 1024;   // H * Dh
constexpr int ATTN_DH = 64;
constexpr int BP = 72;          // bf16 tile pitch (144B rows, 16B aligned)

constexpr int QW_OFF = 0;                  // 64 x BP  (q + r_w_bias)
constexpr int QR_OFF = QW_OFF + 64 * BP;   // 64 x BP  (q + r_r_bias)
constexpr int KT_OFF = QR_OFF + 64 * BP;   // 64 x BP  k tile   [key][d]
constexpr int VT_OFF = KT_OFF + 64 * BP;   // 64 x BP  v tile transposed [d][key]
constexpr int RP_OFF = VT_OFF + 64 * BP;   // 128 x BP rpos window [t][d]
constexpr int PB_OFF = RP_OFF + 128 * BP;  // 64 x BP  probabilities [q][key]
constexpr int BF16_CNT = PB_OFF + 64 * BP;
constexpr int SB_OFF = 0;                  // 64 x 68  combined scores
constexpr int S2_OFF = SB_OFF + 64 * 68;   // 64 x 132 rel scores (t axis)
constexpr int RM_OFF = S2_OFF + 64 * 132;  // running row max
constexpr int RL_OFF = RM_OFF + 64;        // running row sum
constexpr int AL_OFF = RL_OFF + 64;        // rescale factors
constexpr int RED_OFF = AL_OFF + 64;       // 64 x 4 reduction scratch
constexpr int F32_CNT = RED_OFF + 256;
constexpr size_t ATTN_LDS_BYTES = (size_t)BF16_CNT * 2 + (size_t)F32_CNT * 4;

__global__ __launch_bounds__(256)
void relattn_fwd(const __bf16* __restrict__ qbuf, const __bf16* __restrict__ kbuf,
                 const __bf16* __restrict__ vbuf, const __bf16* __restrict__ rpbuf,
                 const float* __restrict__ rwb,  const float* __restrict__ rrb,
                 float* __restrict__ xbuf) {
  extern __shared__ char smem[];
  __bf16* bfb = (__bf16*)smem;
  float*  flb = (float*)(smem + (size_t)BF16_CNT * 2);
  __bf16* qw  = bfb + QW_OFF;
  __bf16* qr  = bfb + QR_OFF;
  __bf16* kt  = bfb + KT_OFF;
  __bf16* vtT = bfb + VT_OFF;
  __bf16* rpt = bfb + RP_OFF;
  __bf16* pbf = bfb + PB_OFF;
  float* sbuf  = flb + SB_OFF;
  float* s2buf = flb + S2_OFF;
  float* rowM  = flb + RM_OFF;
  float* rowL  = flb + RL_OFF;
  float* alpha = flb + AL_OFF;
  float* red   = flb + RED_OFF;

  const int q0 = blockIdx.x * 64;
  const int h  = blockIdx.y;
  const int b  = blockIdx.z;
  const int wave = threadIdx.x >> 5;
  const int lane = threadIdx.x & 31;
  const int ln   = lane & 15;
  const int lh   = lane >> 4;

  const __bf16* qB = qbuf + (size_t)b * ATTN_S * ATTN_HD + h * ATTN_DH;
  const __bf16* kB = kbuf + (size_t)b * ATTN_S * ATTN_HD + h * ATTN_DH;
  const __bf16* vB = vbuf + (size_t)b * ATTN_S * ATTN_HD + h * ATTN_DH;

  // Stage q tile once, with both f32 biases added, back to bf16.
  for (int idx = threadIdx.x; idx < 64 * 64; idx += 256) {
    int r = idx >> 6, c = idx & 63;
    float qv = (float)qB[(size_t)(q0 + r) * ATTN_HD + c];
    qw[r * BP + c] = to_bf16(qv + rwb[h * ATTN_DH + c]);
    qr[r * BP + c] = to_bf16(qv + rrb[h * ATTN_DH + c]);
  }
  for (int idx = threadIdx.x; idx < 64; idx += 256) { rowM[idx] = -1e30f; rowL[idx] = 0.0f; }

  const int mi  = wave & 3;        // wave's O tiles: (mi, ni0), (mi, ni0+2)
  const int ni0 = wave >> 2;
  v8f acc0 = {}, acc1 = {};
  __syncthreads();

  for (int j0 = 0; j0 < ATTN_S; j0 += 64) {
    const int base = (j0 - q0 + ATTN_S - 64) & (ATTN_S - 1); // rpos row for t=0
#if HAS_TDM
    // --- TDM: k tile + rolled rpos window (<=2 descriptors), wave 0 only ---
    if (wave == 0) {
      tdm_load_2d_bf16(lds_addr_of(kt), kB + (size_t)j0 * ATTN_HD,
                       ATTN_HD, ATTN_S, 64, 64, ATTN_HD);
      unsigned n1 = (unsigned)(ATTN_S - base); if (n1 > 128u) n1 = 128u;
      tdm_load_2d_bf16(lds_addr_of(rpt), rpbuf + (size_t)base * ATTN_HD + h * ATTN_DH,
                       ATTN_HD, ATTN_S, 64, n1, ATTN_HD);
      if (n1 < 128u)
        tdm_load_2d_bf16(lds_addr_of(rpt + n1 * BP), rpbuf + h * ATTN_DH,
                         ATTN_HD, ATTN_S, 64, 128u - n1, ATTN_HD);
    }
#else
    for (int idx = threadIdx.x; idx < 64 * 64; idx += 256) {
      int r = idx >> 6, c = idx & 63;
      kt[r * BP + c] = kB[(size_t)(j0 + r) * ATTN_HD + c];
    }
    for (int idx = threadIdx.x; idx < 128 * 64; idx += 256) {
      int r = idx >> 6, c = idx & 63;
      int src = (base + r) & (ATTN_S - 1);
      rpt[r * BP + c] = rpbuf[(size_t)src * ATTN_HD + h * ATTN_DH + c];
    }
#endif
    // --- v^T staged manually (transpose) ----------------------------------
    for (int idx = threadIdx.x; idx < 64 * 64; idx += 256) {
      int r = idx >> 6, c = idx & 63;
      vtT[c * BP + r] = vB[(size_t)(j0 + r) * ATTN_HD + c];
    }
#if HAS_TDM
    if (wave == 0) __builtin_amdgcn_s_wait_tensorcnt(0);
#endif
    __syncthreads();

    // --- s2 = (q + r_r_bias) . rpos^T : 64x128 (32 tiles, 4 per wave) -----
    for (int t = wave * 4; t < wave * 4 + 4; ++t) {
      int tm = t & 3, tn = t >> 2;
      v8f c2 = {};
#pragma unroll
      for (int kk = 0; kk < 64; kk += 32) {
        v16bf a  = lds_frag_A(qr + tm * 16 * BP, BP, kk);
        v16bf bb = lds_frag_BT(rpt, BP, tn * 16, kk);
        c2 = WMMA_BF16(a, bb, c2);
      }
#pragma unroll
      for (int r = 0; r < 8; ++r)
        s2buf[(tm * 16 + r + lh * 8) * 132 + tn * 16 + ln] = c2[r];
    }
    // --- s1 = (q + r_w_bias) . k^T : 64x64 (16 tiles, 2 per wave, regs) ---
    v8f c1[2];
#pragma unroll
    for (int u = 0; u < 2; ++u) {
      int t = wave * 2 + u;
      int tm = t & 3, tn = t >> 2;
      v8f c = {};
#pragma unroll
      for (int kk = 0; kk < 64; kk += 32) {
        v16bf a  = lds_frag_A(qw + tm * 16 * BP, BP, kk);
        v16bf bb = lds_frag_BT(kt, BP, tn * 16, kk);
        c = WMMA_BF16(a, bb, c);
      }
      c1[u] = c;
    }
    __syncthreads();   // s2buf complete

    // --- combine with Transformer-XL shift: t = ji - qi + 63 --------------
#pragma unroll
    for (int u = 0; u < 2; ++u) {
      int t = wave * 2 + u;
      int tm = t & 3, tn = t >> 2;
#pragma unroll
      for (int r = 0; r < 8; ++r) {
        int qi = tm * 16 + r + lh * 8;
        int ji = tn * 16 + ln;
        sbuf[qi * 68 + ji] = (c1[u][r] + s2buf[qi * 132 + ji - qi + 63]) * 0.125f;
      }
    }
    __syncthreads();

    // --- online softmax: 4 threads per row, 16 cols each ------------------
    const int row = threadIdx.x >> 2;
    const int sub = threadIdx.x & 3;
    float sv[16];
    float mloc = -1e30f;
#pragma unroll
    for (int i = 0; i < 16; ++i) {
      sv[i] = sbuf[row * 68 + sub * 16 + i];
      mloc = fmaxf(mloc, sv[i]);
    }
    red[row * 4 + sub] = mloc;
    __syncthreads();
    float mtile = fmaxf(fmaxf(red[row * 4 + 0], red[row * 4 + 1]),
                        fmaxf(red[row * 4 + 2], red[row * 4 + 3]));
    float mold = rowM[row];
    float mnew = fmaxf(mold, mtile);
    float sloc = 0.0f;
#pragma unroll
    for (int i = 0; i < 16; ++i) {
      float p = __expf(sv[i] - mnew);
      sloc += p;
      pbf[row * BP + sub * 16 + i] = to_bf16(p);
    }
    __syncthreads();   // all reads of max done before red reuse
    red[row * 4 + sub] = sloc;
    __syncthreads();
    if (sub == 0) {
      float s4 = red[row * 4] + red[row * 4 + 1] + red[row * 4 + 2] + red[row * 4 + 3];
      float a  = __expf(mold - mnew);
      rowL[row]  = rowL[row] * a + s4;
      rowM[row]  = mnew;
      alpha[row] = a;
    }
    __syncthreads();

    // --- O = O*alpha + P . V ---------------------------------------------
#pragma unroll
    for (int r = 0; r < 8; ++r) {
      float a = alpha[mi * 16 + r + lh * 8];
      acc0[r] *= a;
      acc1[r] *= a;
    }
#pragma unroll
    for (int kk = 0; kk < 64; kk += 32) {
      v16bf ap = lds_frag_A(pbf + mi * 16 * BP, BP, kk);
      v16bf b0 = lds_frag_BT(vtT, BP, ni0 * 16, kk);
      v16bf b1 = lds_frag_BT(vtT, BP, (ni0 + 2) * 16, kk);
      acc0 = WMMA_BF16(ap, b0, acc0);
      acc1 = WMMA_BF16(ap, b1, acc1);
    }
    __syncthreads();   // protect tiles before next iteration restages them
  }

  // --- finalize: divide by row sums, store x[b, q, h, d] -------------------
  float* xB = xbuf + (size_t)b * ATTN_S * ATTN_HD + h * ATTN_DH;
#pragma unroll
  for (int r = 0; r < 8; ++r) {
    int qi = mi * 16 + r + lh * 8;
    float inv = 1.0f / rowL[qi];
    xB[(size_t)(q0 + qi) * ATTN_HD + ni0 * 16 + ln]       = acc0[r] * inv;
    xB[(size_t)(q0 + qi) * ATTN_HD + (ni0 + 2) * 16 + ln] = acc1[r] * inv;
  }
}

// ---------------------------------------------------------------------------
// Host launcher
// ---------------------------------------------------------------------------
extern "C" void kernel_launch(void* const* d_in, const int* in_sizes, int n_in,
                              void* d_out, int out_size, void* d_ws, size_t ws_size,
                              hipStream_t stream) {
  (void)in_sizes; (void)n_in; (void)out_size; (void)ws_size;
  const float* inputs_q  = (const float*)d_in[0];
  const float* inputs_kv = (const float*)d_in[1];
  const float* pos_embed = (const float*)d_in[2];
  const float* Wq   = (const float*)d_in[3];
  const float* bq   = (const float*)d_in[4];
  const float* Wk   = (const float*)d_in[5];
  const float* bk   = (const float*)d_in[6];
  const float* Wv   = (const float*)d_in[7];
  const float* bv   = (const float*)d_in[8];
  const float* Wpos = (const float*)d_in[9];
  const float* r_r_bias = (const float*)d_in[10];
  const float* r_w_bias = (const float*)d_in[11];
  const float* Wout = (const float*)d_in[12];
  const float* bout = (const float*)d_in[13];
  float* out = (float*)d_out;

  const int BS = 2 * 2048;   // B*S rows
  const int D  = 1024;
  const int HD = 1024;       // H*Dh
  // workspace: q, k, v [BS,HD] bf16 ; rpos [S,HD] bf16 ; x [BS,HD] f32
  __bf16* qb = (__bf16*)d_ws;
  __bf16* kb = qb + (size_t)BS * HD;
  __bf16* vb = kb + (size_t)BS * HD;
  __bf16* rp = vb + (size_t)BS * HD;
  float*  xb = (float*)(rp + (size_t)2048 * HD);

  dim3 blk(256);
  // QKV + positional projections (bf16 WMMA GEMMs, bf16 outputs)
  gemm_bf16_wmma<__bf16><<<dim3(BS / 32, HD / 128), blk, 0, stream>>>(inputs_q,  Wq, bq, qb, BS, HD, D);
  gemm_bf16_wmma<__bf16><<<dim3(BS / 32, HD / 128), blk, 0, stream>>>(inputs_kv, Wk, bk, kb, BS, HD, D);
  gemm_bf16_wmma<__bf16><<<dim3(BS / 32, HD / 128), blk, 0, stream>>>(inputs_kv, Wv, bv, vb, BS, HD, D);
  gemm_bf16_wmma<__bf16><<<dim3(2048 / 32, HD / 128), blk, 0, stream>>>(pos_embed, Wpos, nullptr, rp, 2048, HD, D);

  // Fused relative attention (TDM-staged tiles when available)
  (void)hipFuncSetAttribute((const void*)relattn_fwd,
                            hipFuncAttributeMaxDynamicSharedMemorySize,
                            (int)ATTN_LDS_BYTES);
  relattn_fwd<<<dim3(2048 / 64, 16, 2), blk, ATTN_LDS_BYTES, stream>>>(
      qb, kb, vb, rp, r_w_bias, r_r_bias, xb);

  // Output projection (f32 output)
  gemm_bf16_wmma<float><<<dim3(BS / 32, HD / 128), blk, 0, stream>>>(xb, Wout, bout, out, BS, HD, HD);
}
